// CascadeMVSNet_5858335392006
// MI455X (gfx1250) — matrix operand order repaired and checked by reading, not compiled, not run on hardware
//
#include <hip/hip_runtime.h>
#include <hip/hip_bf16.h>

typedef __attribute__((ext_vector_type(16))) _Float16 v16h;
typedef __attribute__((ext_vector_type(8)))  float    v8f;

#define Vn 5
#define Cn 32
#define Hn 128
#define Wn 160
#define Dn 48
#define Gn 8
#define CGn 4
#define HWn (Hn*Wn)
#define NPOS (Dn*Hn*Wn)
#define KTOT 216          /* 27 taps * 8 channels */
#define KT 7              /* ceil(216/32) k-tiles (K padded to 224) */

/* padded, channel-fastest volumes: [z 0..49][y 0..129][x 0..161][ci 0..7] */
#define Dp 50
#define Hp 130
#define Wp 162
#define PADVOL (Dp*Hp*Wp*Gn)          /* 8,424,000 floats */

__host__ __device__ constexpr int tapOff(int t) {
  /* offset (floats) of tap t=(kd,kh,kw) in the padded channel-fastest volume */
  return (((t/9)*Hp + ((t%9)/3))*Wp + (t%3))*Gn;
}

// --------------------------------------------------------------------------
// Kernel 1: projection algebra (single thread).
// mats[(v-1)*12 + {0..8}] = rot row-major, {9,10,11} = trans for
// proj_v = make_proj(projs[v]) @ inv(make_proj(projs[0])).
// --------------------------------------------------------------------------
__global__ void k_proj(const float* __restrict__ projs, float* __restrict__ mats)
{
  if (threadIdx.x != 0 || blockIdx.x != 0) return;
  float P[Vn][4][4];
  for (int v = 0; v < Vn; ++v) {
    const float* E = projs + (v*2 + 0)*16;
    const float* K = projs + (v*2 + 1)*16;
    for (int r = 0; r < 4; ++r)
      for (int c = 0; c < 4; ++c) P[v][r][c] = E[r*4+c];
    for (int r = 0; r < 3; ++r)
      for (int c = 0; c < 4; ++c) {
        float s = 0.f;
        for (int j = 0; j < 3; ++j) s += K[r*4+j]*E[j*4+c];
        P[v][r][c] = s;
      }
  }
  float A[4][8];
  for (int i = 0; i < 4; ++i)
    for (int j = 0; j < 4; ++j) { A[i][j] = P[0][i][j]; A[i][4+j] = (i==j) ? 1.f : 0.f; }
  for (int col = 0; col < 4; ++col) {
    int piv = col; float best = fabsf(A[col][col]);
    for (int r = col+1; r < 4; ++r) { float m = fabsf(A[r][col]); if (m > best) { best = m; piv = r; } }
    if (piv != col)
      for (int j = 0; j < 8; ++j) { float t = A[col][j]; A[col][j] = A[piv][j]; A[piv][j] = t; }
    float inv = 1.f / A[col][col];
    for (int j = 0; j < 8; ++j) A[col][j] *= inv;
    for (int r = 0; r < 4; ++r) if (r != col) {
      float f = A[r][col];
      for (int j = 0; j < 8; ++j) A[r][j] -= f*A[col][j];
    }
  }
  for (int v = 1; v < Vn; ++v) {
    float Pr[4][4];
    for (int r = 0; r < 4; ++r)
      for (int c = 0; c < 4; ++c) {
        float s = 0.f;
        for (int j = 0; j < 4; ++j) s += P[v][r][j]*A[j][4+c];
        Pr[r][c] = s;
      }
    float* m = mats + (v-1)*12;
    for (int r = 0; r < 3; ++r)
      for (int c = 0; c < 3; ++c) m[r*3+c] = Pr[r][c];
    for (int r = 0; r < 3; ++r) m[9+r] = Pr[r][3];
  }
}

// --------------------------------------------------------------------------
// Kernel 2: pack conv1 weights into WMMA A-operand layout.
// GEMM K-order: k = tap*8 + ci  (tap=0..26, ci=0..7), K padded 216->224.
// A[m][k] = w1[m*216 + ci*27 + tap]   (m<8), else 0.
// A layout used (16-bit 16x32, k = 16*half + e): lane<16: e0-7 -> K 0-7, e8-15 -> K 8-15;
//                              lane>=16: e0-7 -> K 16-23, e8-15 -> K 24-31.
// --------------------------------------------------------------------------
__global__ void k_pack_w1(const float* __restrict__ w1, _Float16* __restrict__ Apk)
{
  int t = blockIdx.x*blockDim.x + threadIdx.x;
  if (t >= KT*32) return;
  int kt = t >> 5, lane = t & 31;
  int m = lane & 15;
  int hiLane = lane >> 4;
  #pragma unroll
  for (int e = 0; e < 16; ++e) {
    int quad = hiLane*2 + (e >> 3);
    int k    = kt*32 + quad*8 + (e & 7);
    float v  = 0.f;
    if (m < 8 && k < KTOT) {
      int tap = k >> 3, ci = k & 7;
      v = w1[m*KTOT + ci*27 + tap];
    }
    Apk[(kt*32 + lane)*16 + e] = (_Float16)v;
  }
}

// --------------------------------------------------------------------------
// Kernel 3: transpose feats to channel-fastest layout featsT[v][p][c].
// --------------------------------------------------------------------------
__global__ void k_transpose(const float* __restrict__ feats, float* __restrict__ featsT)
{
  int t = blockIdx.x*blockDim.x + threadIdx.x;
  if (t >= Vn*HWn) return;
  int v = t / HWn;
  int p = t - v*HWn;
  const float* src = feats + (size_t)v*Cn*HWn + p;
  float* dst = featsT + (size_t)t*Cn;
  #pragma unroll
  for (int c = 0; c < Cn; ++c) dst[c] = src[c*HWn];
}

// --------------------------------------------------------------------------
// Kernel 4: zero both padded volumes (halo must be zero).
// --------------------------------------------------------------------------
__global__ void k_zero(float4* __restrict__ ptr, int n4)
{
  int i = blockIdx.x*blockDim.x + threadIdx.x;
  if (i < n4) ptr[i] = make_float4(0.f, 0.f, 0.f, 0.f);
}

// --------------------------------------------------------------------------
// Kernel 5: group-correlation cost volume with bilinear homography warp.
// Writes padded channel-fastest volp at (d+1, h+1, w+1).
// --------------------------------------------------------------------------
__global__ void k_costvol(const float* __restrict__ featsT,
                          const float* __restrict__ mats,
                          const float* __restrict__ depth_values,
                          float* __restrict__ volp)
{
  int t = blockIdx.x*blockDim.x + threadIdx.x;
  if (t >= NPOS) return;
  int d = t / HWn;
  int p = t - d*HWn;
  int h = p / Wn, w = p - h*Wn;
  float dep = depth_values[d];

  const float4* rp = (const float4*)(featsT + (size_t)p*Cn);  // view 0
  float4 ref4[8];
  #pragma unroll
  for (int j = 0; j < 8; ++j) ref4[j] = rp[j];

  float acc[Gn];
  #pragma unroll
  for (int g = 0; g < Gn; ++g) acc[g] = 0.f;

  float wf = (float)w, hf = (float)h;
  for (int v = 1; v < Vn; ++v) {
    const float* M = mats + (v-1)*12;
    float fxr = M[0]*wf + M[1]*hf + M[2];
    float fyr = M[3]*wf + M[4]*hf + M[5];
    float fzr = M[6]*wf + M[7]*hf + M[8];
    float z   = fzr*dep + M[11];
    float px  = (fxr*dep + M[9])  / z;
    float py  = (fyr*dep + M[10]) / z;

    float fx0 = floorf(px), fy0 = floorf(py);
    float fx1 = fx0 + 1.f,  fy1 = fy0 + 1.f;
    float wx1 = px - fx0, wx0 = 1.f - wx1;
    float wy1 = py - fy0, wy0 = 1.f - wy1;

    float vx0 = (fx0 >= 0.f && fx0 <= (float)(Wn-1)) ? 1.f : 0.f;
    float vx1 = (fx1 >= 0.f && fx1 <= (float)(Wn-1)) ? 1.f : 0.f;
    float vy0 = (fy0 >= 0.f && fy0 <= (float)(Hn-1)) ? 1.f : 0.f;
    float vy1 = (fy1 >= 0.f && fy1 <= (float)(Hn-1)) ? 1.f : 0.f;

    int cx0 = (int)fminf(fmaxf(fx0, 0.f), (float)(Wn-1));
    int cx1 = (int)fminf(fmaxf(fx1, 0.f), (float)(Wn-1));
    int cy0 = (int)fminf(fmaxf(fy0, 0.f), (float)(Hn-1));
    int cy1 = (int)fminf(fmaxf(fy1, 0.f), (float)(Hn-1));

    float w00 = wx0*wy0*vx0*vy0;
    float w01 = wx1*wy0*vx1*vy0;
    float w10 = wx0*wy1*vx0*vy1;
    float w11 = wx1*wy1*vx1*vy1;

    const float* fb = featsT + (size_t)v*HWn*Cn;
    const float4* q00 = (const float4*)(fb + (size_t)(cy0*Wn + cx0)*Cn);
    const float4* q01 = (const float4*)(fb + (size_t)(cy0*Wn + cx1)*Cn);
    const float4* q10 = (const float4*)(fb + (size_t)(cy1*Wn + cx0)*Cn);
    const float4* q11 = (const float4*)(fb + (size_t)(cy1*Wn + cx1)*Cn);

    #pragma unroll
    for (int j = 0; j < 8; ++j) {      // float4 j == channel group j (CG=4)
      float4 a = q00[j], b = q01[j], c = q10[j], e = q11[j];
      float sx = w00*a.x + w01*b.x + w10*c.x + w11*e.x;
      float sy = w00*a.y + w01*b.y + w10*c.y + w11*e.y;
      float sz = w00*a.z + w01*b.z + w10*c.z + w11*e.z;
      float sw2 = w00*a.w + w01*b.w + w10*c.w + w11*e.w;
      acc[j] += ref4[j].x*sx + ref4[j].y*sy + ref4[j].z*sz + ref4[j].w*sw2;
    }
  }
  const float scale = 1.f / (float)(CGn*(Vn-1));   // 1/16
  int ob = (((d+1)*Hp + (h+1))*Wp + (w+1))*Gn;
  float4 o0 = make_float4(acc[0]*scale, acc[1]*scale, acc[2]*scale, acc[3]*scale);
  float4 o1 = make_float4(acc[4]*scale, acc[5]*scale, acc[6]*scale, acc[7]*scale);
  *(float4*)(volp + ob)     = o0;
  *(float4*)(volp + ob + 4) = o1;
}

// --------------------------------------------------------------------------
// Kernel 6: conv1 (8->8, 3x3x3) + bias + ReLU as WMMA GEMM over K=224.
// B operand: element e of lane <-> (tap, ci=e&7); two taps per lane per
// k-tile -> 4 x b128 contiguous loads from the padded volume. Tap offsets
// are compile-time constants selected by lane-half. A staged in LDS.
// --------------------------------------------------------------------------
__global__ void k_conv1_wmma(const float* __restrict__ volp,
                             const _Float16* __restrict__ Apk,
                             const float* __restrict__ b1,
                             float* __restrict__ cost1p)
{
  __shared__ v16h sA[KT*32];
  {
    const float4* Ag = (const float4*)Apk;
    float4* As = (float4*)sA;
    for (int i = threadIdx.x; i < KT*32*2; i += blockDim.x) As[i] = Ag[i];
  }
  __syncthreads();

  int lane = threadIdx.x & 31;
  int wave = blockIdx.x*(blockDim.x >> 5) + (threadIdx.x >> 5);
  int n    = wave*16 + (lane & 15);
  int dcol = n / HWn;
  int rem  = n - dcol*HWn;
  int hcol = rem / Wn;
  int wcol = rem - hcol*Wn;
  bool hi  = (lane & 16) != 0;
  int base = ((dcol*Hp + hcol)*Wp + wcol)*Gn;   // padded coords: +kd,+kh,+kw via tapOff

  v8f acc = {};
  #pragma unroll
  for (int kt = 0; kt < KT; ++kt) {
    v16h a = sA[kt*32 + lane];
    int oA = hi ? tapOff(4*kt + 2) : tapOff(4*kt);
    int oB = hi ? tapOff((4*kt + 3 > 26) ? 26 : 4*kt + 3) : tapOff(4*kt + 1);
    const float4* pA = (const float4*)(volp + base + oA);
    const float4* pB = (const float4*)(volp + base + oB);
    float4 fa0 = pA[0], fa1 = pA[1];
    float4 fb0 = pB[0], fb1 = pB[1];
    v16h b;
    b[0]=(_Float16)fa0.x;  b[1]=(_Float16)fa0.y;  b[2]=(_Float16)fa0.z;  b[3]=(_Float16)fa0.w;
    b[4]=(_Float16)fa1.x;  b[5]=(_Float16)fa1.y;  b[6]=(_Float16)fa1.z;  b[7]=(_Float16)fa1.w;
    b[8]=(_Float16)fb0.x;  b[9]=(_Float16)fb0.y;  b[10]=(_Float16)fb0.z; b[11]=(_Float16)fb0.w;
    b[12]=(_Float16)fb1.x; b[13]=(_Float16)fb1.y; b[14]=(_Float16)fb1.z; b[15]=(_Float16)fb1.w;
    acc = __builtin_amdgcn_wmma_f32_16x16x32_f16(
        false, a, false, b, (short)0, acc, false, false);
  }
  // C layout: VGPR r -> M = r + 8*(lane>=16); only lanes<16 hold real rows.
  if (lane < 16) {
    int ob = (((dcol+1)*Hp + (hcol+1))*Wp + (wcol+1))*Gn;
    float4 s0 = make_float4(fmaxf(acc[0]+b1[0],0.f), fmaxf(acc[1]+b1[1],0.f),
                            fmaxf(acc[2]+b1[2],0.f), fmaxf(acc[3]+b1[3],0.f));
    float4 s1 = make_float4(fmaxf(acc[4]+b1[4],0.f), fmaxf(acc[5]+b1[5],0.f),
                            fmaxf(acc[6]+b1[6],0.f), fmaxf(acc[7]+b1[7],0.f));
    *(float4*)(cost1p + ob)     = s0;
    *(float4*)(cost1p + ob + 4) = s1;
  }
}

// --------------------------------------------------------------------------
// Kernel 7: conv2 (8->1, 3x3x3) + bias on padded channel-fastest cost1p.
// Weights transposed [tap][ci] into LDS; no bounds checks.
// --------------------------------------------------------------------------
__global__ void k_conv2(const float* __restrict__ cost1p,
                        const float* __restrict__ w2,
                        const float* __restrict__ b2,
                        float* __restrict__ cost2)
{
  __shared__ float sw[KTOT];                 // [tap*8 + ci]
  for (int i = threadIdx.x; i < KTOT; i += blockDim.x) {
    int tap = i >> 3, ci = i & 7;
    sw[i] = w2[ci*27 + tap];
  }
  __syncthreads();

  int t = blockIdx.x*blockDim.x + threadIdx.x;
  if (t >= NPOS) return;
  int d = t / HWn;
  int p = t - d*HWn;
  int h = p / Wn, w = p - h*Wn;
  int base = ((d*Hp + h)*Wp + w)*Gn;

  float sum = b2[0];
  #pragma unroll
  for (int tap = 0; tap < 27; ++tap) {
    const float4* q = (const float4*)(cost1p + base + tapOff(tap));
    float4 u0 = q[0], u1 = q[1];
    sum += sw[tap*8+0]*u0.x + sw[tap*8+1]*u0.y + sw[tap*8+2]*u0.z + sw[tap*8+3]*u0.w
         + sw[tap*8+4]*u1.x + sw[tap*8+5]*u1.y + sw[tap*8+6]*u1.z + sw[tap*8+7]*u1.w;
  }
  cost2[t] = sum;
}

// --------------------------------------------------------------------------
// Kernel 8: per-pixel softmax over D, expected depth, 4-tap confidence.
// cost2 lives in the prob region of d_out; prob written back in place.
// --------------------------------------------------------------------------
__global__ void k_softmax(const float* __restrict__ depth_values,
                          float* __restrict__ out)
{
  int p = blockIdx.x*blockDim.x + threadIdx.x;
  if (p >= HWn) return;
  float* prob = out + 2*HWn;

  float c[Dn];
  #pragma unroll
  for (int d = 0; d < Dn; ++d) c[d] = prob[d*HWn + p];

  float m = c[0];
  #pragma unroll
  for (int d = 1; d < Dn; ++d) m = fmaxf(m, c[d]);

  float s = 0.f;
  #pragma unroll
  for (int d = 0; d < Dn; ++d) { c[d] = __expf(c[d] - m); s += c[d]; }
  float inv = 1.f / s;

  float depth = 0.f, fidx = 0.f;
  #pragma unroll
  for (int d = 0; d < Dn; ++d) {
    c[d] *= inv;
    depth += c[d]*depth_values[d];
    fidx  += c[d]*(float)d;
  }
  int idx = (int)fidx;
  idx = max(0, min(Dn-1, idx));

  float conf = 0.f;
  #pragma unroll
  for (int d = 0; d < Dn; ++d)
    if (d >= idx-1 && d <= idx+2) conf += c[d];

  #pragma unroll
  for (int d = 0; d < Dn; ++d) prob[d*HWn + p] = c[d];
  out[p]       = depth;
  out[HWn + p] = conf;
}

// --------------------------------------------------------------------------
extern "C" void kernel_launch(void* const* d_in, const int* in_sizes, int n_in,
                              void* d_out, int out_size, void* d_ws, size_t ws_size,
                              hipStream_t stream)
{
  (void)in_sizes; (void)n_in; (void)out_size; (void)ws_size;
  const float* feats  = (const float*)d_in[0];
  const float* projs  = (const float*)d_in[1];
  const float* depthv = (const float*)d_in[2];
  const float* w1     = (const float*)d_in[3];
  const float* b1     = (const float*)d_in[4];
  const float* w2     = (const float*)d_in[5];
  const float* b2     = (const float*)d_in[6];
  float* out = (float*)d_out;

  float*    ws     = (float*)d_ws;
  float*    mats   = ws;                                  // 48 floats
  _Float16* Apk    = (_Float16*)(ws + 64);                // 7*32*16 halves
  float*    featsT = ws + 64 + 1792;                      // 5*20480*32 floats
  float*    volp   = featsT + (size_t)Vn*HWn*Cn;          // padded volume
  float*    cost1p = volp + (size_t)PADVOL;               // padded volume
  float*    cost2  = out + 2*HWn;                         // reuse prob region

  const int n4zero = (2*PADVOL)/4;

  k_proj      <<<1, 1, 0, stream>>>(projs, mats);
  k_pack_w1   <<<1, 256, 0, stream>>>(w1, Apk);
  k_transpose <<<(Vn*HWn)/256, 256, 0, stream>>>(feats, featsT);
  k_zero      <<<(n4zero + 255)/256, 256, 0, stream>>>((float4*)volp, n4zero);
  k_costvol   <<<NPOS/256, 256, 0, stream>>>(featsT, mats, depthv, volp);
  k_conv1_wmma<<<(NPOS/16)/8, 256, 0, stream>>>(volp, Apk, b1, cost1p);
  k_conv2     <<<NPOS/256, 256, 0, stream>>>(cost1p, w2, b2, cost2);
  k_softmax   <<<(HWn + 255)/256, 256, 0, stream>>>(depthv, out);
}
